// GDNAttnBackend_10118942949426
// MI455X (gfx1250) — matrix operand order repaired and checked by reading, not compiled, not run on hardware
//
#include <hip/hip_runtime.h>
#include <hip/hip_bf16.h>

#define T_LEN    2048
#define CONV_D   4096
#define KEY_DIM  1024
#define NUM_KH   16
#define NUM_VH   32
#define CHUNK    16
#define NCHUNK   (T_LEN / CHUNK)

typedef __attribute__((ext_vector_type(16))) __bf16 bf16x16;
typedef __attribute__((ext_vector_type(8)))  float  f32x8;

static __device__ inline f32x8 wmma_bf16(bf16x16 a, bf16x16 b, f32x8 c) {
  // D = A(16x32 bf16) * B(32x16 bf16) + C(16x16 f32)
  return __builtin_amdgcn_wmma_f32_16x16x32_bf16(false, a, false, b, (short)0, c, false, false);
}

// K-index inside a 16-bit A(16x32) operand: element e of the v16bf, lane half hi.
static __device__ inline int a_kidx(int e, int hi) {
  return hi * 16 + e;
}

// ---------------------------------------------------------------------------
// Kernel 1: causal depthwise conv1d (W=4) + SiLU, split q/k/v, L2-norm q/k.
// q scaled by 64^-0.5.  q/k stored bf16 (WMMA operands), v stored f32.
// Grid: (T, 64) blocks of 64 threads; block (t, s) handles 64 channels.
// ---------------------------------------------------------------------------
__global__ __launch_bounds__(64) void gdn_prep(const float* __restrict__ x,
                                               const float* __restrict__ cstate,
                                               const float* __restrict__ w,
                                               __bf16* __restrict__ wsq,
                                               __bf16* __restrict__ wsk,
                                               float*  __restrict__ wsv) {
  const int t = blockIdx.x, s = blockIdx.y, tid = threadIdx.x;
  const int d = s * 64 + tid;
  float acc = 0.f;
#pragma unroll
  for (int wi = 0; wi < 4; ++wi) {
    int r = t + wi;  // padded row index; rows 0..2 come from conv_state
    float val = (r < 3) ? cstate[r * CONV_D + d] : x[(r - 3) * CONV_D + d];
    acc += w[d * 4 + wi] * val;
  }
  float y = acc / (1.f + __expf(-acc));   // SiLU

  __shared__ float red[64];
  if (s < 32) {                            // q or k slice: L2 norm over 64 channels
    red[tid] = y * y;
    __syncthreads();
    for (int o = 32; o > 0; o >>= 1) {
      if (tid < o) red[tid] += red[tid + o];
      __syncthreads();
    }
    float rn = rsqrtf(red[0] + 1e-6f);
    if (s < 16) wsq[t * KEY_DIM + d] = (__bf16)(y * rn * 0.125f);
    else        wsk[t * KEY_DIM + (d - KEY_DIM)] = (__bf16)(y * rn);
  } else {                                 // v slice
    wsv[t * 2048 + (d - 2048)] = y;
  }
}

// ---------------------------------------------------------------------------
// Kernel 2: gates  g = -exp(A_log)*softplus(a+dt_bias),  beta = sigmoid(b)
// ---------------------------------------------------------------------------
__global__ void gdn_gates(const float* __restrict__ b, const float* __restrict__ a,
                          const float* __restrict__ A_log, const float* __restrict__ dtb,
                          float* __restrict__ g, float* __restrict__ beta) {
  int idx = blockIdx.x * blockDim.x + threadIdx.x;
  if (idx >= T_LEN * NUM_VH) return;
  int hh = idx & 31;
  float av = a[idx] + dtb[hh];
  float sp = (av > 20.f) ? av : log1pf(__expf(av));
  g[idx]    = -__expf(A_log[hh]) * sp;
  beta[idx] = 1.f / (1.f + __expf(-b[idx]));
}

// ---------------------------------------------------------------------------
// Kernel 3: chunked gated delta-rule scan, chunk C = 16 (== WMMA tile).
// One wave per (head, dv-half): block = 32 threads, grid = 64 blocks.
// Per chunk (all WMMA bf16, fp32 accumulate; state S fp32-resident in LDS):
//   KS0 = K*S0, QS0 = Q*S0, KK = K*K^T, QK = Q*K^T
//   A[i,j] = beta_i * exp(cg_i - cg_j) * KK[i,j]   (j <  i)
//   B[i,j] =          exp(cg_i - cg_j) * QK[i,j]   (j <= i)
//   rhs_i  = beta_i * (v_i - gam_i * KS0_i);  u = (I+A)^-1 rhs  (fwd subst)
//   o_i    = gam_i * QS0_i + (B*U)_i
//   S      = gam_C * S + Khat^T * U,  Khat_j = exp(cg_C - cg_j) * k_j
// ---------------------------------------------------------------------------
__global__ __launch_bounds__(32) void gdn_scan(const __bf16* __restrict__ wsq,
                                               const __bf16* __restrict__ wsk,
                                               const float*  __restrict__ wsv,
                                               const float*  __restrict__ wsg,
                                               const float*  __restrict__ wsb,
                                               const float*  __restrict__ s_init,
                                               float* __restrict__ out) {
  const int h    = blockIdx.x >> 1;
  const int half = blockIdx.x & 1;     // which 32-wide dv slice
  const int kh   = h >> 1;             // q/k head (rep = 2)
  const int lane = threadIdx.x;
  const int lo = lane & 15, hi = lane >> 4;

  __shared__ float  S[64 * 32];                    // state [dk=64][dv_local=32], fp32
  __shared__ __bf16 Kc[16 * 64], Qc[16 * 64], Khat[16 * 64];
  __shared__ __bf16 Ub[16 * 32];
  __shared__ float  Uf[16 * 32];
  __shared__ float  Aatt[16 * 16], Batt[16 * 16];
  __shared__ float  cg[16], gam[16], bet[16], dex[16], gamC;

  for (int i = lane; i < 64 * 32; i += 32)
    S[i] = s_init[h * 4096 + (i >> 5) * 64 + half * 32 + (i & 31)];
  __syncthreads();

  for (int ch = 0; ch < NCHUNK; ++ch) {
    const int t0 = ch * CHUNK;

    // ---- P0: stage chunk operands ----
    for (int i = lane; i < 16 * 64; i += 32) {
      int r = i >> 6, c = i & 63;
      Qc[i] = wsq[((t0 + r) * NUM_KH + kh) * 64 + c];
      Kc[i] = wsk[((t0 + r) * NUM_KH + kh) * 64 + c];
    }
    if (lane == 0) {
      float run = 0.f;
      for (int i = 0; i < 16; ++i) {
        run += wsg[(t0 + i) * NUM_VH + h];
        cg[i]  = run;
        gam[i] = __expf(run);
        bet[i] = wsb[(t0 + i) * NUM_VH + h];
      }
      for (int j = 0; j < 16; ++j) dex[j] = __expf(run - cg[j]);
      gamC = __expf(run);
    }
    __syncthreads();

    // ---- P1: WMMA front matmuls ----
    bf16x16 ak0, ak1, aq0, aq1;
#pragma unroll
    for (int e = 0; e < 16; ++e) {
      int kk = a_kidx(e, hi);
      ak0[e] = Kc[lo * 64 + kk];       ak1[e] = Kc[lo * 64 + 32 + kk];
      aq0[e] = Qc[lo * 64 + kk];       aq1[e] = Qc[lo * 64 + 32 + kk];
    }

    f32x8 ks0[2], qs0[2];
#pragma unroll
    for (int nt = 0; nt < 2; ++nt) {
      bf16x16 b0, b1;
#pragma unroll
      for (int e = 0; e < 16; ++e) {
        int kd = hi * 16 + e;
        b0[e] = (__bf16)S[kd * 32 + nt * 16 + lo];
        b1[e] = (__bf16)S[(kd + 32) * 32 + nt * 16 + lo];
      }
      f32x8 c0 = {}; c0 = wmma_bf16(ak0, b0, c0); c0 = wmma_bf16(ak1, b1, c0); ks0[nt] = c0;
      f32x8 c1 = {}; c1 = wmma_bf16(aq0, b0, c1); c1 = wmma_bf16(aq1, b1, c1); qs0[nt] = c1;
    }

    bf16x16 bt0, bt1;                   // B = K^T (B[k,n] = Kc[n][k])
#pragma unroll
    for (int e = 0; e < 16; ++e) {
      int kd = hi * 16 + e;
      bt0[e] = Kc[lo * 64 + kd];
      bt1[e] = Kc[lo * 64 + 32 + kd];
    }
    f32x8 kkap = {}; kkap = wmma_bf16(ak0, bt0, kkap); kkap = wmma_bf16(ak1, bt1, kkap);
    f32x8 qkap = {}; qkap = wmma_bf16(aq0, bt0, qkap); qkap = wmma_bf16(aq1, bt1, qkap);

#pragma unroll
    for (int j = 0; j < 8; ++j) {       // decay/causal masking of attention mats
      int m = hi * 8 + j, n = lo;
      Aatt[m * 16 + n] = (n < m)  ? bet[m] * __expf(cg[m] - cg[n]) * kkap[j] : 0.f;
      Batt[m * 16 + n] = (n <= m) ?          __expf(cg[m] - cg[n]) * qkap[j] : 0.f;
    }
#pragma unroll
    for (int nt = 0; nt < 2; ++nt)      // rhs -> Uf
#pragma unroll
      for (int j = 0; j < 8; ++j) {
        int m = hi * 8 + j, c = nt * 16 + lo;
        float vval = wsv[((t0 + m) * NUM_VH + h) * 64 + half * 32 + c];
        Uf[m * 32 + c] = bet[m] * (vval - gam[m] * ks0[nt][j]);
      }
    __syncthreads();

    // ---- P2: forward substitution (lane-private column, no barriers) ----
    {
      const int c = lane;
      for (int i = 1; i < 16; ++i) {
        float acc = 0.f;
        for (int j = 0; j < i; ++j) acc += Aatt[i * 16 + j] * Uf[j * 32 + c];
        Uf[i * 32 + c] -= acc;
      }
    }
    __syncthreads();

    // ---- P3: bf16 working copies ----
    for (int i = lane; i < 16 * 32; i += 32) Ub[i] = (__bf16)Uf[i];
    for (int i = lane; i < 16 * 64; i += 32) Khat[i] = (__bf16)(dex[i >> 6] * (float)Kc[i]);
    __syncthreads();

    bf16x16 bu[2];                      // B = U (k = 16, zero-padded to 32)
#pragma unroll
    for (int nt = 0; nt < 2; ++nt)
#pragma unroll
      for (int e = 0; e < 16; ++e)
        bu[nt][e] = (hi == 0) ? Ub[e * 32 + nt * 16 + lo] : (__bf16)0.f;

    // ---- P4: O = gam_i * QS0 + Batt * U ----
    bf16x16 ab;
#pragma unroll
    for (int e = 0; e < 16; ++e) {
      int kk = a_kidx(e, hi);
      ab[e] = (kk < 16) ? (__bf16)Batt[lo * 16 + kk] : (__bf16)0.f;
    }
#pragma unroll
    for (int nt = 0; nt < 2; ++nt) {
      f32x8 o2 = {}; o2 = wmma_bf16(ab, bu[nt], o2);
#pragma unroll
      for (int j = 0; j < 8; ++j) {
        int m = hi * 8 + j;
        out[((t0 + m) * NUM_VH + h) * 64 + half * 32 + nt * 16 + lo] =
            gam[m] * qs0[nt][j] + o2[j];
      }
    }

    // ---- P5: state update  S = gamC*S + Khat^T * U ----
    float gC = gamC;
#pragma unroll
    for (int mt = 0; mt < 4; ++mt) {
      bf16x16 at;                       // A = Khat^T (rows = dk, k = 16 padded)
#pragma unroll
      for (int e = 0; e < 16; ++e) {
        int kk = a_kidx(e, hi);
        at[e] = (kk < 16) ? Khat[kk * 64 + mt * 16 + lo] : (__bf16)0.f;
      }
#pragma unroll
      for (int nt = 0; nt < 2; ++nt) {
        f32x8 c;
#pragma unroll
        for (int j = 0; j < 8; ++j)
          c[j] = gC * S[(mt * 16 + hi * 8 + j) * 32 + nt * 16 + lo];
        f32x8 sn = wmma_bf16(at, bu[nt], c);
#pragma unroll
        for (int j = 0; j < 8; ++j)
          S[(mt * 16 + hi * 8 + j) * 32 + nt * 16 + lo] = sn[j];
      }
    }
    __syncthreads();
  }
}

// ---------------------------------------------------------------------------
extern "C" void kernel_launch(void* const* d_in, const int* in_sizes, int n_in,
                              void* d_out, int out_size, void* d_ws, size_t ws_size,
                              hipStream_t stream) {
  const float* x      = (const float*)d_in[0];   // mixed_qkv [T,4096]
  const float* cstate = (const float*)d_in[1];   // conv_state_in [1,3,4096]
  const float* s_init = (const float*)d_in[2];   // recurrent_state_in [1,32,64,64]
  const float* b_in   = (const float*)d_in[3];   // b [T,32]
  const float* a_in   = (const float*)d_in[4];   // a [T,32]
  const float* w      = (const float*)d_in[5];   // conv1d_weight [4096,4]
  const float* A_log  = (const float*)d_in[6];   // [32]
  const float* dtb    = (const float*)d_in[7];   // [32]

  char* ws = (char*)d_ws;
  __bf16* wsq = (__bf16*)(ws);                         // 4 MB  q (bf16, normed, scaled)
  __bf16* wsk = (__bf16*)(ws + (4u  << 20));           // 4 MB  k (bf16, normed)
  float*  wsv = (float* )(ws + (8u  << 20));           // 16 MB v (f32)
  float*  wsg = (float* )(ws + (24u << 20));           // 256 KB g
  float*  wsb = (float* )(ws + (24u << 20) + (1u << 19)); // 256 KB beta

  gdn_prep <<<dim3(T_LEN, 64), 64, 0, stream>>>(x, cstate, w, wsq, wsk, wsv);
  gdn_gates<<<(T_LEN * NUM_VH + 255) / 256, 256, 0, stream>>>(b_in, a_in, A_log, dtb, wsg, wsb);
  gdn_scan <<<64, 32, 0, stream>>>(wsq, wsk, wsv, wsg, wsb, s_init, (float*)d_out);
}
